// ConvKernel_24670292148737
// MI455X (gfx1250) — matrix operand order repaired and checked by reading, not compiled, not run on hardware
//
#include <hip/hip_runtime.h>
#include <hip/hip_bf16.h>
#include <math.h>

// ---------------------------------------------------------------------------
// Problem constants (from reference)
// ---------------------------------------------------------------------------
constexpr int kE   = 240000;   // edges
constexpr int kN   = 60000;    // nodes
constexpr int kW   = 256;      // width / width_norm
constexpr int kDH  = 32;       // dim_head
constexpr int kNH  = 8;        // num_head
constexpr int kF   = 128;      // dim_head*4 (per-head act width)

typedef __bf16 bf16_t;
typedef __attribute__((ext_vector_type(8)))  bf16_t bf16x8;
typedef __attribute__((ext_vector_type(16))) bf16_t bf16x16;
typedef __attribute__((ext_vector_type(8)))  float  f32x8;

// ws layout (bf16 element offsets); all 32B-aligned
constexpr size_t kOffLinPre   = 0;        // 256x256
constexpr size_t kOffGateLin  = 65536;    // 256x256
constexpr size_t kOffValueLin = 131072;   // 256x256
constexpr size_t kOffLoraDown = 196608;   // 256x32
constexpr size_t kOffLoraUp   = 204800;   // 32x256
constexpr size_t kOffGateK    = 212992;   // 8 x (32x128)
constexpr size_t kOffValueK   = 245760;   // 8 x (32x128)
constexpr size_t kOffPostK    = 278528;   // 8 x (128x32)
constexpr size_t kBf16Elems   = 311296;   // total bf16 elements in ws
// float tables appended after the bf16 region:
//   tab[0..63]    softplus(moa_w)  (j*32+k)
//   tab[64..127]  softplus(moa_s)
//   tab[128..129] 1/sum_k softplus(moa_w[j])

// ---------------------------------------------------------------------------
// WMMA helpers (CDNA5 wave32, 16x16x32 bf16 -> f32)
// ---------------------------------------------------------------------------
__device__ __forceinline__ f32x8 wmma_bf16(bf16x16 a, bf16x16 b, f32x8 c) {
  // 8 args: (neg_a, A, neg_b, B, c_mod, C, reuse_a, reuse_b)
  return __builtin_amdgcn_wmma_f32_16x16x32_bf16(false, a, false, b,
                                                 (short)0, c, false, false);
}

// A fragment: M=16, K=32 bf16 from LDS.
// Lane L (0..15): row=L, K 0..7 & 16..23.  Lane L+16: row=L, K 8..15 & 24..31.
__device__ __forceinline__ bf16x16 load_a_lds(const bf16_t* base, int ldk,
                                              int kt, int lane) {
  int row = lane & 15;
  int kb  = kt * 32 + ((lane >> 4) << 3);
  const bf16_t* p = base + row * ldk + kb;
  bf16x8 lo = *(const bf16x8*)(p);
  bf16x8 hi = *(const bf16x8*)(p + 16);
  return __builtin_shufflevector(lo, hi, 0,1,2,3,4,5,6,7,
                                         8,9,10,11,12,13,14,15);
}

// B fragment from PRE-SWIZZLED weights: 32 contiguous bytes per lane
// (two global_load_b128, no packing VALU).
__device__ __forceinline__ bf16x16 load_b_sw(const bf16_t* __restrict__ w,
                                             int NT, int kt, int nt, int lane) {
  const bf16_t* p = w + (((size_t)(kt * NT + nt)) << 9) + (lane << 4);
  bf16x8 lo = *(const bf16x8*)(p);
  bf16x8 hi = *(const bf16x8*)(p + 8);
  return __builtin_shufflevector(lo, hi, 0,1,2,3,4,5,6,7,
                                         8,9,10,11,12,13,14,15);
}

__device__ __forceinline__ float softplus_f(float x) {
  return (x > 20.f) ? x : log1pf(expf(x));
}

// Scatter one f32 KxN weight element into the swizzled bf16 fragment layout.
// dst[i] with i = (kt*NT + nt)*512 + lane*16 + e  <-  src[k*N + col]
__device__ __forceinline__ void swz(const float* __restrict__ src,
                                    bf16_t* __restrict__ dst, int N, int i) {
  int e    = i & 15;
  int lane = (i >> 4) & 31;
  int tile = i >> 9;
  int NT   = N >> 4;
  int kt   = tile / NT;
  int nt   = tile - kt * NT;
  int col  = nt * 16 + (lane & 15);
  int k    = kt * 32 + ((lane >> 4) << 3) + (e & 7) + ((e >> 3) << 4);
  dst[i] = (bf16_t)src[(size_t)k * N + col];
}

// ---------------------------------------------------------------------------
// Kernel 0: swizzle+downconvert weights into WMMA fragment layout, and
// precompute the MoAct softplus tables (identical across all edges/blocks).
// ---------------------------------------------------------------------------
__global__ __launch_bounds__(256)
void prep_kernel(const float* __restrict__ lin_pre,
                 const float* __restrict__ gate_lin,
                 const float* __restrict__ value_lin,
                 const float* __restrict__ lora_down,
                 const float* __restrict__ lora_up,
                 const float* __restrict__ gate_k,
                 const float* __restrict__ value_k,
                 const float* __restrict__ post_k,
                 const float* __restrict__ moa_w,
                 const float* __restrict__ moa_s,
                 bf16_t* __restrict__ ws) {
  int i = blockIdx.x * 256 + threadIdx.x;
  if (i < 65536) {
    swz(lin_pre,   ws + kOffLinPre,   kW, i);
    swz(gate_lin,  ws + kOffGateLin,  kW, i);
    swz(value_lin, ws + kOffValueLin, kW, i);
  }
  if (i < 32768) {
    int h = i >> 12, li = i & 4095;                    // 4096 elems per head
    swz(gate_k  + h * kDH * kF, ws + kOffGateK  + h * 4096, kF,  li);
    swz(value_k + h * kDH * kF, ws + kOffValueK + h * 4096, kF,  li);
    swz(post_k  + h * kF * kDH, ws + kOffPostK  + h * 4096, kDH, li);
  }
  if (i < 8192) {
    swz(lora_down, ws + kOffLoraDown, kDH, i);
    swz(lora_up,   ws + kOffLoraUp,   kW,  i);
  }
  float* tab = (float*)(ws + kBf16Elems);
  if (i < 64) {
    tab[i]      = softplus_f(moa_w[i]);
    tab[64 + i] = softplus_f(moa_s[i]);
  }
  if (i < 2) {
    float s = 0.f;
#pragma unroll 1
    for (int k = 0; k < kDH; ++k) s += softplus_f(moa_w[i * kDH + k]);
    tab[128 + i] = 1.f / s;
  }
}

// ---------------------------------------------------------------------------
// Kernel 1: edge messages.  16 edges per block, 256 threads = 8 waves.
// msg@lin_pre + ((msg@lora_down)*emb)@lora_up, atomic scatter-add to agg.
// ---------------------------------------------------------------------------
__global__ __launch_bounds__(256)
void edge_kernel(const float* __restrict__ x,
                 const int*   __restrict__ edge_idx,
                 const int*   __restrict__ edge_attr,
                 const float* __restrict__ node_elec,
                 const float* __restrict__ emb_edge,
                 const float* __restrict__ elec_lin,
                 const float* __restrict__ moa_tab,
                 const bf16_t* __restrict__ lin_pre_bf,
                 const bf16_t* __restrict__ lora_down_bf,
                 const bf16_t* __restrict__ lora_up_bf,
                 float* __restrict__ agg) {
  __shared__ int s_i0[16], s_i1[16];
  __shared__ __align__(16) bf16_t s_msg[16][kW];      // 8 KB
  __shared__ float s_moa[16][2];
  __shared__ float s_emb[16][kDH];
  __shared__ float s_lmid[16][kDH];
  __shared__ __align__(16) bf16_t s_lora[16][kDH];

  const int t     = threadIdx.x;
  const int lane  = t & 31;
  const int wid   = t >> 5;
  const int ebase = blockIdx.x * 16;

  if (t < 16) {
    s_i0[t] = edge_idx[ebase + t];
    s_i1[t] = edge_idx[kE + ebase + t];
  }
  __syncthreads();

  // gather msg = x[i0] + x[i1] -> bf16 LDS (coalesced per edge row)
#pragma unroll
  for (int i = 0; i < 16; ++i) {
    float v = x[(size_t)s_i0[i] * kW + t] + x[(size_t)s_i1[i] * kW + t];
    s_msg[i][t] = (bf16_t)v;
  }

  // tanh-mixture electronegativity term (threads 0..15, one edge each);
  // softplus tables are precomputed -> uniform scalar loads + 64 tanh.
  if (t < 16) {
    float d0 = node_elec[s_i0[t] * 2 + 0] - node_elec[s_i1[t] * 2 + 0];
    float d1 = node_elec[s_i0[t] * 2 + 1] - node_elec[s_i1[t] * 2 + 1];
    float acc0 = 0.f, acc1 = 0.f;
#pragma unroll 1
    for (int k = 0; k < kDH; ++k) {
      acc0 += tanhf(d0 * moa_tab[64 + k])       * moa_tab[k];
      acc1 += tanhf(d1 * moa_tab[64 + kDH + k]) * moa_tab[kDH + k];
    }
    s_moa[t][0] = acc0 * moa_tab[128];
    s_moa[t][1] = acc1 * moa_tab[129];
  }
  __syncthreads();

  // edge embedding: bond-table sum + moa @ elec_lin
#pragma unroll
  for (int i = 0; i < 2; ++i) {
    int idx = i * 256 + t;            // 16*32 = 512 values
    int e = idx >> 5, d = idx & 31;
    int a0 = edge_attr[(ebase + e) * 3 + 0];
    int a1 = edge_attr[(ebase + e) * 3 + 1];
    int a2 = edge_attr[(ebase + e) * 3 + 2];
    s_emb[e][d] = emb_edge[a0 * kDH + d] + emb_edge[a1 * kDH + d] +
                  emb_edge[a2 * kDH + d] +
                  s_moa[e][0] * elec_lin[d] + s_moa[e][1] * elec_lin[kDH + d];
  }

  // lora_mid = msg @ lora_down   [16,32], waves 0..1 (one 16-col tile each)
  if (wid < 2) {
    f32x8 c = {};
#pragma unroll
    for (int kt = 0; kt < 8; ++kt) {
      bf16x16 a = load_a_lds(&s_msg[0][0], kW, kt, lane);
      bf16x16 b = load_b_sw(lora_down_bf, 2, kt, wid, lane);
      c = wmma_bf16(a, b, c);
    }
    int col = wid * 16 + (lane & 15);
#pragma unroll
    for (int r = 0; r < 8; ++r)
      s_lmid[r + ((lane >> 4) << 3)][col] = c[r];
  }
  __syncthreads();

  // modulate by emb -> bf16 A operand for lora_up
#pragma unroll
  for (int i = 0; i < 2; ++i) {
    int idx = i * 256 + t;
    int e = idx >> 5, d = idx & 31;
    s_lora[e][d] = (bf16_t)(s_lmid[e][d] * s_emb[e][d]);
  }
  __syncthreads();

  // hoist A fragments (invariant across N tiles)
  bf16x16 aa[8];
#pragma unroll
  for (int kt = 0; kt < 8; ++kt)
    aa[kt] = load_a_lds(&s_msg[0][0], kW, kt, lane);
  bf16x16 alora = load_a_lds(&s_lora[0][0], kDH, 0, lane);

  // out tile = msg @ lin_pre + lora' @ lora_up; atomic scatter to agg[ei1]
#pragma unroll
  for (int n = 0; n < 2; ++n) {
    int nt = wid * 2 + n;                         // 8 waves x 2 = 16 col tiles
    f32x8 c = {};
#pragma unroll
    for (int kt = 0; kt < 8; ++kt) {
      bf16x16 b = load_b_sw(lin_pre_bf, 16, kt, nt, lane);
      c = wmma_bf16(aa[kt], b, c);
    }
    {
      bf16x16 b = load_b_sw(lora_up_bf, 16, 0, nt, lane);
      c = wmma_bf16(alora, b, c);
    }
    int col = nt * 16 + (lane & 15);
#pragma unroll
    for (int r = 0; r < 8; ++r) {
      int row = r + ((lane >> 4) << 3);
      atomicAdd(&agg[(size_t)s_i1[row] * kW + col], c[r]);
    }
  }
}

// ---------------------------------------------------------------------------
// Kernel 2: node post-block.  16 nodes per block, 256 threads = 8 waves.
// Reads agg tile from d_out, writes final output in place (row n -> row n).
// ---------------------------------------------------------------------------
__global__ __launch_bounds__(256)
void node_kernel(float* __restrict__ inout,         // agg in, result out
                 const int*   __restrict__ deg,
                 const float* __restrict__ emb_deg,
                 const float* __restrict__ act_bias,
                 const bf16_t* __restrict__ gate_lin_bf,
                 const bf16_t* __restrict__ value_lin_bf,
                 const bf16_t* __restrict__ gate_k_bf,
                 const bf16_t* __restrict__ value_k_bf,
                 const bf16_t* __restrict__ post_k_bf) {
  __shared__ __align__(16) bf16_t s_agg[16][kW];    // 8 KB
  __shared__ float s_preG[16][kW];                  // 16 KB
  __shared__ float s_preV[16][kW];                  // 16 KB
  __shared__ __align__(16) bf16_t s_nG[16][kW];     // 8 KB
  __shared__ __align__(16) bf16_t s_nV[16][kW];     // 8 KB
  __shared__ __align__(16) bf16_t s_xx[kNH][16][kF];// 32 KB

  const int t     = threadIdx.x;
  const int lane  = t & 31;
  const int wid   = t >> 5;
  const int nbase = blockIdx.x * 16;

#pragma unroll
  for (int i = 0; i < 16; ++i)
    s_agg[i][t] = (bf16_t)inout[(size_t)(nbase + i) * kW + t];
  __syncthreads();

  // hoist A fragments of agg (shared by all 4 tiles this wave computes)
  bf16x16 aa[8];
#pragma unroll
  for (int kt = 0; kt < 8; ++kt)
    aa[kt] = load_a_lds(&s_agg[0][0], kW, kt, lane);

  // preG = agg@gate_lin, preV = agg@value_lin : 32 col tiles over 8 waves
#pragma unroll
  for (int j = 0; j < 4; ++j) {
    int T   = wid * 4 + j;
    int mat = T >> 4;                    // 0 = gate, 1 = value
    int nt  = T & 15;
    const bf16_t* wb = mat ? value_lin_bf : gate_lin_bf;
    float* dst = mat ? &s_preV[0][0] : &s_preG[0][0];
    f32x8 c = {};
#pragma unroll
    for (int kt = 0; kt < 8; ++kt) {
      bf16x16 b = load_b_sw(wb, 16, kt, nt, lane);
      c = wmma_bf16(aa[kt], b, c);
    }
    int col = nt * 16 + (lane & 15);
#pragma unroll
    for (int r = 0; r < 8; ++r)
      dst[(r + ((lane >> 4) << 3)) * kW + col] = c[r];
  }
  __syncthreads();

  // per-(node, head) RMS norm; gate adds degree-embedding bias then /sqrt2
  if (t < 128) {
    int e = t >> 3, h = t & 7;
    int dg = deg[nbase + e];
    float sg = 0.f, sv = 0.f;
    for (int d = 0; d < kDH; ++d) {
      float g = s_preG[e][h * kDH + d];  sg += g * g;
      float v = s_preV[e][h * kDH + d];  sv += v * v;
    }
    float rg = rsqrtf(sg * (1.f / kDH) + 1e-6f);
    float rv = rsqrtf(sv * (1.f / kDH) + 1e-6f);
    const float inv_s2 = 0.70710678118654752f;
    for (int d = 0; d < kDH; ++d) {
      float bias = emb_deg[(size_t)dg * kW + h * kDH + d];
      s_nG[e][h * kDH + d] = (bf16_t)((s_preG[e][h * kDH + d] * rg + bias) * inv_s2);
      s_nV[e][h * kDH + d] = (bf16_t)(s_preV[e][h * kDH + d] * rv);
    }
  }
  __syncthreads();

  // wave w owns head w: gate/value 32->128 GEMMs + softplus gate
  {
    int h = wid;
    bf16x16 ag = load_a_lds(&s_nG[0][h * kDH], kW, 0, lane);
    bf16x16 av = load_a_lds(&s_nV[0][h * kDH], kW, 0, lane);
#pragma unroll
    for (int nt = 0; nt < 8; ++nt) {
      f32x8 cg = {}, cv = {};
      bf16x16 bg = load_b_sw(gate_k_bf  + (size_t)h * 4096, 8, 0, nt, lane);
      cg = wmma_bf16(ag, bg, cg);
      bf16x16 bv = load_b_sw(value_k_bf + (size_t)h * 4096, 8, 0, nt, lane);
      cv = wmma_bf16(av, bv, cv);
      int col = nt * 16 + (lane & 15);
#pragma unroll
      for (int r = 0; r < 8; ++r) {
        int row = r + ((lane >> 4) << 3);
        float g = cg[r] + act_bias[h * kF + col];
        float a = softplus_f(g);
        a = fminf(fmaxf(a, 0.22360679775f), 4.47213595500f);  // [1/sqrt20, sqrt20]
        s_xx[h][row][col] = (bf16_t)(a * cv[r]);
      }
    }
  }
  __syncthreads();

  // post GEMM per head: [16,128] @ [128,32] -> write output in place
  {
    int h = wid;
    bf16x16 ax[4];
#pragma unroll
    for (int kt = 0; kt < 4; ++kt)
      ax[kt] = load_a_lds(&s_xx[h][0][0], kF, kt, lane);
#pragma unroll
    for (int nt = 0; nt < 2; ++nt) {
      f32x8 c = {};
#pragma unroll
      for (int kt = 0; kt < 4; ++kt) {
        bf16x16 b = load_b_sw(post_k_bf + (size_t)h * 4096, 2, kt, nt, lane);
        c = wmma_bf16(ax[kt], b, c);
      }
      int col = nt * 16 + (lane & 15);
#pragma unroll
      for (int r = 0; r < 8; ++r) {
        int row = r + ((lane >> 4) << 3);
        inout[(size_t)(nbase + row) * kW + h * kDH + col] = c[r];
      }
    }
  }
}

// ---------------------------------------------------------------------------
// Launch
// ---------------------------------------------------------------------------
extern "C" void kernel_launch(void* const* d_in, const int* in_sizes, int n_in,
                              void* d_out, int out_size, void* d_ws,
                              size_t ws_size, hipStream_t stream) {
  const float* x         = (const float*)d_in[0];
  const int*   deg       = (const int*)  d_in[1];
  const int*   edge_idx  = (const int*)  d_in[2];
  const int*   edge_attr = (const int*)  d_in[3];
  const float* node_elec = (const float*)d_in[4];
  const float* lora_down = (const float*)d_in[5];
  const float* lora_up   = (const float*)d_in[6];
  const float* emb_edge  = (const float*)d_in[7];
  const float* moa_w     = (const float*)d_in[8];
  const float* moa_s     = (const float*)d_in[9];
  const float* elec_lin  = (const float*)d_in[10];
  const float* emb_deg   = (const float*)d_in[11];
  const float* lin_pre   = (const float*)d_in[12];
  const float* gate_lin  = (const float*)d_in[13];
  const float* gate_k    = (const float*)d_in[14];
  const float* value_lin = (const float*)d_in[15];
  const float* value_k   = (const float*)d_in[16];
  const float* act_bias  = (const float*)d_in[17];
  const float* post_k    = (const float*)d_in[18];

  float*  out = (float*)d_out;
  bf16_t* wsb = (bf16_t*)d_ws;
  const float* moa_tab = (const float*)(wsb + kBf16Elems);

  // d_out doubles as agg (segment_sum target); node kernel is in-place.
  hipMemsetAsync(d_out, 0, (size_t)kN * kW * sizeof(float), stream);

  prep_kernel<<<256, 256, 0, stream>>>(lin_pre, gate_lin, value_lin, lora_down,
                                       lora_up, gate_k, value_k, post_k,
                                       moa_w, moa_s, wsb);

  edge_kernel<<<kE / 16, 256, 0, stream>>>(
      x, edge_idx, edge_attr, node_elec, emb_edge, elec_lin, moa_tab,
      wsb + kOffLinPre, wsb + kOffLoraDown, wsb + kOffLoraUp, out);

  node_kernel<<<kN / 16, 256, 0, stream>>>(
      out, deg, emb_deg, act_bias, wsb + kOffGateLin, wsb + kOffValueLin,
      wsb + kOffGateK, wsb + kOffValueK, wsb + kOffPostK);
}
